// Tiresias_3650722201713
// MI455X (gfx1250) — matrix operand order repaired and checked by reading, compiled, not run on hardware
//
#include <hip/hip_runtime.h>
#include <hip/hip_bf16.h>

// ---------------------------------------------------------------------------
// SoftArrayLSTM forward for MI455X (gfx1250, wave32, WMMA).
//   INPUT=1024, H=256, OUT=1024, K=4 lanes, B=64, T=128, GATE=4*H*K=4096
// Strategy:
//   prep_kernel : repack Wh -> bf16 WMMA B-fragments with gate permutation
//                 p = j*16 + (k*4+g); fold biases; zero h ping-pong; init bar.
//   lstm_kernel : persistent, 16 WGs x 256 thr (8 waves). Each WG owns 16
//                 hidden units (256 permuted gate rows). Per step:
//                   - attention softmax (redundant per WG, tiny)
//                   - pack h -> bf16 A-fragments in LDS
//                   - WMMA GEMM (M=64,K=256,N=256 slice), fused epilogue:
//                     bias + Wi one-hot gather + LSTM cell + attn*max-pool
//                   - grid-wide sense barrier
//   out_kernel  : logits = h @ Wl^T + bl, log_softmax.
// ---------------------------------------------------------------------------

#define H_    256
#define KL    4
#define BB    64
#define TT    128
#define GATE  4096
#define VOCAB 1024
#define OUTD  1024
#define NWG   16
#define TPB   256
#define JPW   16   // hidden units per workgroup

typedef __attribute__((ext_vector_type(16))) __bf16 v16bf;
typedef __attribute__((ext_vector_type(8)))  float  v8f;

__device__ __forceinline__ float fsig(float x)  { return 1.0f / (1.0f + __expf(-x)); }
__device__ __forceinline__ float ftanh(float x) { return 1.0f - 2.0f / (__expf(2.0f * x) + 1.0f); }

// permuted gate index p = j*16 + (k*4+g)  ->  original row k*1024 + g*256 + j
__device__ __forceinline__ int orig_row(int p) {
    int j = p >> 4, kg = p & 15, k = kg >> 2, g = kg & 3;
    return k * 1024 + g * 256 + j;
}

// ---------------------------------------------------------------------------
__global__ void __launch_bounds__(TPB)
prep_kernel(const float* __restrict__ Wh, const float* __restrict__ bi,
            const float* __restrict__ bh, __bf16* __restrict__ whp,
            float* __restrict__ bias_p, float* __restrict__ hbuf,
            unsigned* __restrict__ bar) {
    int gid = blockIdx.x * blockDim.x + threadIdx.x;   // 0 .. 1048575
    // Pack Wh into B-fragment tile order: tile = j*8 + kt, 512 bf16/tile,
    // element q = lane*16 + slot.  B layout (32x16 KxN, 16-bit):
    //   n = lane&15 (gate component kg), K = (slot>>3)*16 + (lane>>4)*8 + (slot&7).
    {
        int tile = gid >> 9;
        int q    = gid & 511;
        int j    = tile >> 3, kt = tile & 7;
        int lane = q >> 4, slot = q & 15;
        int n     = lane & 15;
        int khalf = lane >> 4;
        int kk    = kt * 32 + (slot >> 3) * 16 + khalf * 8 + (slot & 7);
        int row   = orig_row(j * 16 + n);
        whp[gid]  = (__bf16)Wh[row * H_ + kk];
    }
    if (gid < GATE) {
        int r = orig_row(gid);
        bias_p[gid] = bi[r] + bh[r];
    }
    if (gid < 2 * BB * H_) hbuf[gid] = 0.0f;
    if (gid == 0) { bar[0] = 0u; bar[1] = 0u; }
}

// ---------------------------------------------------------------------------
__global__ void __launch_bounds__(TPB)
lstm_kernel(const int* __restrict__ X, const float* __restrict__ Wi,
            const float* __restrict__ Wa, const float* __restrict__ ba,
            const __bf16* __restrict__ whp, const float* __restrict__ bias_p,
            float* __restrict__ hbuf, unsigned* __restrict__ bar) {
    __shared__ __attribute__((aligned(16))) __bf16 hA[32 * 512]; // 32KB A-frags
    __shared__ float gate_s[8 * 256];          // per-wave 16x16 f32 staging
    __shared__ float c_sl[BB * KL * JPW];      // persistent cell-state slice
    __shared__ float attn_s[BB * KL];
    __shared__ float bias_l[256];

    const int tid  = threadIdx.x;
    const int wg   = blockIdx.x;
    const int wv   = tid >> 5;
    const int lane = tid & 31;

    bias_l[tid] = bias_p[wg * 256 + tid];
    for (int e = tid; e < BB * KL * JPW; e += TPB) c_sl[e] = 0.0f;
    __syncthreads();

    for (int t = 0; t < TT; ++t) {
        const float* __restrict__ hcur = hbuf + (t & 1) * BB * H_;
        float* __restrict__       hnxt = hbuf + ((t + 1) & 1) * BB * H_;

        // ---- raw attention scores: attn_s[b*4+k] = h[b]·Wa[k] + ba[k] ----
        {
            int b = tid >> 2, k = tid & 3;
            const float* hr = hcur + b * H_;
            const float* wr = Wa + k * H_;
            float s = ba[k];
            #pragma unroll 4
            for (int j = 0; j < H_; ++j) s += hr[j] * wr[j];
            attn_s[tid] = s;
        }
        // ---- pack h into bf16 A-fragments (16x32 MxK per tile) ----
        for (int e = tid; e < 32 * 512; e += TPB) {
            int tile = e >> 9, q = e & 511;
            int mt = tile >> 3, kt = tile & 7;
            int ln = q >> 4, slot = q & 15;
            int m = ln & 15, lh = ln >> 4;
            int v = slot >> 1, sl = slot & 1;
            int kk = ((v >> 2) * 16) + lh * 8 + (v & 3) * 2 + sl;
            hA[e] = (__bf16)hcur[(mt * 16 + m) * H_ + kt * 32 + kk];
        }
        __syncthreads();
        // ---- softmax over the 4 lanes ----
        {
            int b = tid >> 2, k = tid & 3;
            float r0 = attn_s[b * 4 + 0], r1 = attn_s[b * 4 + 1];
            float r2 = attn_s[b * 4 + 2], r3 = attn_s[b * 4 + 3];
            __syncthreads();
            float mx = fmaxf(fmaxf(r0, r1), fmaxf(r2, r3));
            float e0 = __expf(r0 - mx), e1 = __expf(r1 - mx);
            float e2 = __expf(r2 - mx), e3 = __expf(r3 - mx);
            float inv = 1.0f / (e0 + e1 + e2 + e3);
            float ek = (k == 0) ? e0 : (k == 1) ? e1 : (k == 2) ? e2 : e3;
            attn_s[tid] = ek * inv;
        }
        __syncthreads();

        // ---- GEMM + fused cell update; wave wv owns jj = wv*2 + {0,1} ----
        for (int ti = 0; ti < 8; ++ti) {
            const int mt    = ti >> 1;
            const int jj    = wv * 2 + (ti & 1);
            const int jglob = wg * JPW + jj;
            if (ti < 7) {  // prefetch next tile row of packed Wh into caches
                int jn = wv * 2 + ((ti + 1) & 1);
                __builtin_prefetch(&whp[(size_t)(wg * JPW + jn) * 8 * 512], 0, 0);
            }
            v8f acc = {};
            #pragma unroll
            for (int kt = 0; kt < 8; ++kt) {
                v16bf afrag = *(const v16bf*)&hA[(mt * 8 + kt) * 512 + lane * 16];
                v16bf bfrag = *(const v16bf*)&whp[(size_t)(jglob * 8 + kt) * 512 + lane * 16];
                acc = __builtin_amdgcn_wmma_f32_16x16x32_bf16(
                          false, afrag, false, bfrag, (short)0, acc, false, false);
            }
            // stage D-tile (rows = batch, cols = 16 gate components of unit jglob)
            {
                float* gs = &gate_s[wv * 256];
                const int m0 = (lane >> 4) * 8;
                const int n  = lane & 15;
                #pragma unroll
                for (int r = 0; r < 8; ++r) gs[(m0 + r) * 16 + n] = acc[r];
            }
            __syncthreads();
            if (lane < 16) {
                const int m   = lane;
                const int b   = mt * 16 + m;
                const int tok = X[b * TT + t];
                const float* gr  = &gate_s[wv * 256 + m * 16];
                const float* wib = Wi + tok;          // Wi[row*VOCAB + tok]
                float hmax = -3.4e38f;
                #pragma unroll
                for (int k = 0; k < KL; ++k) {
                    const int rb = k * 1024 + jglob;
                    float gi = gr[k*4+0] + bias_l[jj*16 + k*4+0] + wib[(size_t)(rb      ) * VOCAB];
                    float gf = gr[k*4+1] + bias_l[jj*16 + k*4+1] + wib[(size_t)(rb + 256) * VOCAB];
                    float gg = gr[k*4+2] + bias_l[jj*16 + k*4+2] + wib[(size_t)(rb + 512) * VOCAB];
                    float go = gr[k*4+3] + bias_l[jj*16 + k*4+3] + wib[(size_t)(rb + 768) * VOCAB];
                    float ig = fsig(gi), fg = fsig(gf), gv = ftanh(gg);
                    float og = fsig(go) * attn_s[b * 4 + k];
                    float c  = c_sl[(b * 4 + k) * JPW + jj];
                    c = fg * c + ig * gv;
                    c_sl[(b * 4 + k) * JPW + jj] = c;
                    hmax = fmaxf(hmax, og * ftanh(c));
                }
                hnxt[b * H_ + jglob] = hmax;
            }
            __syncthreads();
        }

        // ---- grid-wide barrier (all 16 persistent WGs) ----
        __threadfence();
        __syncthreads();
        if (tid == 0) {
            unsigned gen = __atomic_load_n(&bar[1], __ATOMIC_ACQUIRE);
            if (atomicAdd(&bar[0], 1u) == (unsigned)(NWG - 1)) {
                bar[0] = 0u;
                __atomic_store_n(&bar[1], gen + 1u, __ATOMIC_RELEASE);
            } else {
                while (__atomic_load_n(&bar[1], __ATOMIC_ACQUIRE) == gen)
                    __builtin_amdgcn_s_sleep(2);
            }
        }
        __syncthreads();
    }
}

// ---------------------------------------------------------------------------
__global__ void __launch_bounds__(256)
out_kernel(const float* __restrict__ hfin, const float* __restrict__ Wl,
           const float* __restrict__ bl, float* __restrict__ out) {
    __shared__ float hs[H_];
    __shared__ float lg[OUTD];
    __shared__ float red[256];
    const int b = blockIdx.x, tid = threadIdx.x;
    hs[tid] = hfin[b * H_ + tid];
    __syncthreads();
    #pragma unroll
    for (int q = 0; q < 4; ++q) {
        int o = q * 256 + tid;
        const float* wr = Wl + o * H_;
        float s = bl[o];
        #pragma unroll 4
        for (int j = 0; j < H_; ++j) s += hs[j] * wr[j];
        lg[o] = s;
    }
    __syncthreads();
    float lm = fmaxf(fmaxf(lg[tid], lg[256 + tid]), fmaxf(lg[512 + tid], lg[768 + tid]));
    red[tid] = lm;
    __syncthreads();
    for (int s = 128; s > 0; s >>= 1) {
        if (tid < s) red[tid] = fmaxf(red[tid], red[tid + s]);
        __syncthreads();
    }
    float mx = red[0];
    __syncthreads();
    float le = __expf(lg[tid] - mx) + __expf(lg[256 + tid] - mx) +
               __expf(lg[512 + tid] - mx) + __expf(lg[768 + tid] - mx);
    red[tid] = le;
    __syncthreads();
    for (int s = 128; s > 0; s >>= 1) {
        if (tid < s) red[tid] += red[tid + s];
        __syncthreads();
    }
    float lse = mx + __logf(red[0]);
    #pragma unroll
    for (int q = 0; q < 4; ++q) {
        int o = q * 256 + tid;
        out[b * OUTD + o] = lg[o] - lse;
    }
}

// ---------------------------------------------------------------------------
extern "C" void kernel_launch(void* const* d_in, const int* in_sizes, int n_in,
                              void* d_out, int out_size, void* d_ws, size_t ws_size,
                              hipStream_t stream) {
    (void)in_sizes; (void)n_in; (void)out_size; (void)ws_size;
    const int*   X  = (const int*)  d_in[0];
    const float* Wi = (const float*)d_in[1];
    const float* bi = (const float*)d_in[2];
    const float* Wh = (const float*)d_in[3];
    const float* bh = (const float*)d_in[4];
    const float* Wa = (const float*)d_in[5];
    const float* ba = (const float*)d_in[6];
    const float* Wl = (const float*)d_in[7];
    const float* bl = (const float*)d_in[8];
    float* out = (float*)d_out;

    // workspace layout
    char* ws = (char*)d_ws;
    __bf16*   whp    = (__bf16*)ws;                              // 2 MB packed Wh
    float*    bias_p = (float*)(ws + 2097152);                   // 16 KB
    float*    hbuf   = (float*)(ws + 2097152 + 16384);           // 128 KB ping-pong h
    unsigned* bar    = (unsigned*)(ws + 2097152 + 16384 + 131072);

    prep_kernel<<<GATE * H_ / TPB, TPB, 0, stream>>>(Wh, bi, bh, whp, bias_p, hbuf, bar);
    lstm_kernel<<<NWG, TPB, 0, stream>>>(X, Wi, Wa, ba, whp, bias_p, hbuf, bar);
    // T=128 is even -> final h lives in hbuf parity 0
    out_kernel<<<BB, 256, 0, stream>>>(hbuf, Wl, bl, out);
}
